// AttentionHead_5214090297398
// MI455X (gfx1250) — matrix operand order repaired and checked by reading, not compiled, not run on hardware
//
#include <hip/hip_runtime.h>
#include <stddef.h>

// ---------------------------------------------------------------------------
// AttentionHead (no softmax) on gfx1250:
//   out = sqrt(128) * ( (x@Wq^T+bq) @ ( (x@Wk^T+bk)^T @ (x@Wv^T+bv) ) )
// Reassociated to avoid the 8192x8192 scores matrix entirely.
// Heavy math: v_wmma_f32_16x16x32_bf16 (f32 accumulation).
// Shared-operand staging: global_load_async_to_lds_b128 + s_wait_asynccnt.
// ---------------------------------------------------------------------------

#define NROWS   8192
#define EMBDIM  1024
#define DDIM    128
#define KSPLIT  32          // split-K workgroups for M = k^T v

typedef __attribute__((ext_vector_type(16))) __bf16 bf16x16;
typedef __attribute__((ext_vector_type(8)))  float  floatx8;
typedef int v4i_vec __attribute__((vector_size(16)));   // matches builtin param type

struct __attribute__((aligned(16))) U128 { unsigned int w[4]; };
struct __attribute__((aligned(16))) U256 { U128 lo, hi; };
union  Frag  { U256 u; bf16x16 v; };

__device__ __forceinline__ unsigned short f2bf(float f) {
    unsigned int u = __float_as_uint(f);
    u += 0x7FFFu + ((u >> 16) & 1u);          // round-to-nearest-even
    return (unsigned short)(u >> 16);
}

__device__ __forceinline__ floatx8 wmma_bf16(const Frag& a, const Frag& b, floatx8 c) {
    // (neg_a, A, neg_b, B, c_mod, C, reuse_a, reuse_b)
    return __builtin_amdgcn_wmma_f32_16x16x32_bf16(false, a.v, false, b.v,
                                                   (short)0, c, false, false);
}

// --------------------------- async global->LDS -----------------------------
#if __has_builtin(__builtin_amdgcn_global_load_async_to_lds_b128)
#define ASYNC_LDS_OK 1
#endif

__device__ __forceinline__ void async_copy16(const unsigned short* g, unsigned short* l) {
#ifdef ASYNC_LDS_OK
    __builtin_amdgcn_global_load_async_to_lds_b128((v4i_vec*)g, (v4i_vec*)l, 0, 0);
#else
    *(U128*)l = *(const U128*)g;               // sync fallback
#endif
}

__device__ __forceinline__ void wait_async_le2() {
#if __has_builtin(__builtin_amdgcn_s_wait_asynccnt)
    __builtin_amdgcn_s_wait_asynccnt(2);
#else
    asm volatile("s_wait_asynccnt 0x2" ::: "memory");
#endif
}
__device__ __forceinline__ void wait_async_0() {
#if __has_builtin(__builtin_amdgcn_s_wait_asynccnt)
    __builtin_amdgcn_s_wait_asynccnt(0);
#else
    asm volatile("s_wait_asynccnt 0x0" ::: "memory");
#endif
}

// ---------------------------------------------------------------------------
// Kernel 1: convert Wq|Wk|Wv (each [128][1024] f32 row-major) to bf16.
// Row-major keeps WMMA B-fragments (W[n][k], contiguous in k) as 32B loads.
// ---------------------------------------------------------------------------
__global__ __launch_bounds__(256) void prep_kernel(const float* __restrict__ Wq,
                                                   const float* __restrict__ Wk,
                                                   const float* __restrict__ Wv,
                                                   unsigned short* __restrict__ Wbf) {
    const int WN = DDIM * EMBDIM;                     // 131072
    int i = blockIdx.x * 256 + threadIdx.x;           // < 3*WN
    if      (i <     WN) Wbf[i] = f2bf(Wq[i]);
    else if (i < 2 * WN) Wbf[i] = f2bf(Wk[i -     WN]);
    else if (i < 3 * WN) Wbf[i] = f2bf(Wv[i - 2 * WN]);
}

// ---------------------------------------------------------------------------
// Kernel 2: fused QKV projection.
// Grid: 512 WGs (16 rows each), 8 waves; wave w owns column tile n0=16*w of
// q, k and v (3 f32 accumulators). x tile staged as bf16 in LDS, read once.
// Outputs: q row-major bf16 [8192][128]; kT, vT bf16 [128][8192] so that the
// downstream WMMA A/B fragments are contiguous aligned loads.
// ---------------------------------------------------------------------------
__global__ __launch_bounds__(256) void qkv_kernel(const float* __restrict__ x,
                                                  const unsigned short* __restrict__ Wbf,
                                                  const float* __restrict__ bq,
                                                  const float* __restrict__ bk,
                                                  const float* __restrict__ bv,
                                                  unsigned short* __restrict__ q,
                                                  unsigned short* __restrict__ kT,
                                                  unsigned short* __restrict__ vT) {
    __shared__ __attribute__((aligned(16))) unsigned short xlds[16][72]; // 64 + pad

    const int m0   = blockIdx.x * 16;
    const int wave = threadIdx.x >> 5;
    const int lane = threadIdx.x & 31;
    const int ln   = lane & 15;
    const int g    = lane >> 4;
    const int n0   = wave * 16;

    floatx8 acc[3];
    #pragma unroll
    for (int s = 0; s < 3; ++s) acc[s] = (floatx8)0.0f;

    for (int k0 = 0; k0 < EMBDIM; k0 += 64) {
        // cooperative load + f32->bf16 convert of x[m0:+16][k0:+64]
        #pragma unroll
        for (int j = 0; j < 4; ++j) {
            int idx = (int)threadIdx.x + j * 256;     // 0..1023
            int r = idx >> 6, c = idx & 63;
            xlds[r][c] = f2bf(x[(size_t)(m0 + r) * EMBDIM + k0 + c]);
        }
        if (k0 + 64 < EMBDIM) {
            // hint next x chunk into cache (global_prefetch_b8)
            __builtin_prefetch(&x[(size_t)(m0 + (threadIdx.x >> 4)) * EMBDIM +
                                  k0 + 64 + (threadIdx.x & 15) * 4], 0, 3);
        }
        __syncthreads();

        #pragma unroll
        for (int s = 0; s < 2; ++s) {                 // two K=32 steps per chunk
            Frag a;                                   // A: lane m, K = g*16..g*16+15
            a.u.lo = *(const U128*)&xlds[ln][s * 32 + g * 16];
            a.u.hi = *(const U128*)&xlds[ln][s * 32 + g * 16 + 8];
            #pragma unroll
            for (int sel = 0; sel < 3; ++sel) {       // q, k, v
                const unsigned short* wrow = Wbf + (size_t)sel * (DDIM * EMBDIM)
                                           + (size_t)(n0 + ln) * EMBDIM
                                           + k0 + s * 32 + g * 16;
                Frag b;                               // B: lane n, K = g*16..g*16+15
                b.u.lo = *(const U128*)wrow;
                b.u.hi = *(const U128*)(wrow + 8);
                acc[sel] = wmma_bf16(a, b, acc[sel]);
            }
        }
        __syncthreads();
    }

    const int   col = n0 + ln;
    const float biq = bq[col], bik = bk[col], biv = bv[col];
    #pragma unroll
    for (int r = 0; r < 8; ++r) {                     // C: row = r + 8*g, col = ln
        int row = m0 + r + g * 8;
        q [(size_t)row * DDIM  + col] = f2bf(acc[0][r] + biq);
        kT[(size_t)col * NROWS + row] = f2bf(acc[1][r] + bik);
        vT[(size_t)col * NROWS + row] = f2bf(acc[2][r] + biv);
    }
}

// ---------------------------------------------------------------------------
// Kernel 3: M = k^T @ v, K = 8192 split over KSPLIT WGs.
// The vT chunk (B operand) is shared by all 8 waves -> stage it in LDS via
// double-buffered GLOBAL_LOAD_ASYNC_TO_LDS_B128 (ASYNCcnt), barrier-published.
// Wave w owns d-tile d0 = 16*w x all 8 e-tiles (8 accumulators).
// Deterministic: each WG writes its own partial [e][d] block (no atomics).
// ---------------------------------------------------------------------------
__device__ __forceinline__ void stage_v(const unsigned short* __restrict__ vT,
                                        int n, unsigned short* buf, int t) {
    // copy vT[0:128][n:n+32] (8 KB) -> buf, 2 x 16B per thread
    #pragma unroll
    for (int h = 0; h < 2; ++h) {
        int idx  = t + h * 256;                       // 0..511 16B-segments
        int e    = idx >> 2;                          // row (stride 8192)
        int col8 = (idx & 3) * 8;                     // element offset in chunk
        async_copy16(vT + (size_t)e * NROWS + n + col8, buf + idx * 8);
    }
}

__global__ __launch_bounds__(256) void kv_kernel(const unsigned short* __restrict__ kT,
                                                 const unsigned short* __restrict__ vT,
                                                 float* __restrict__ Mpart) {
    __shared__ __attribute__((aligned(16))) unsigned short vtile[2][DDIM * 32]; // 2 x 8KB

    const int ROWS_PER_WG = NROWS / KSPLIT;           // 256
    const int NITER       = ROWS_PER_WG / 32;         // 8
    const int nbase = blockIdx.x * ROWS_PER_WG;
    const int wave  = threadIdx.x >> 5;
    const int lane  = threadIdx.x & 31;
    const int ln    = lane & 15;
    const int g     = lane >> 4;
    const int d0    = wave * 16;
    const int t     = (int)threadIdx.x;

    floatx8 acc[8];
    #pragma unroll
    for (int i = 0; i < 8; ++i) acc[i] = (floatx8)0.0f;

    stage_v(vT, nbase, &vtile[0][0], t);              // prologue: buffer 0

    for (int it = 0; it < NITER; ++it) {
        const int n = nbase + it * 32;
        if (it + 1 < NITER) {                         // issue next buffer early
            stage_v(vT, n + 32, &vtile[(it + 1) & 1][0], t);
            wait_async_le2();                         // current buffer's copies done
        } else {
            wait_async_0();
        }
        __syncthreads();                              // publish across waves

        Frag a;                                       // A[d][n] = kT[d][n] (row-major)
        const unsigned short* kp = kT + (size_t)(d0 + ln) * NROWS + n;
        a.u.lo = *(const U128*)(kp + g * 16);
        a.u.hi = *(const U128*)(kp + g * 16 + 8);
        #pragma unroll
        for (int et = 0; et < 8; ++et) {              // B[n][e] = vT[e][n], from LDS
            const unsigned short* bp = &vtile[it & 1][(et * 16 + ln) * 32 + g * 16];
            Frag b;
            b.u.lo = *(const U128*)bp;
            b.u.hi = *(const U128*)(bp + 8);
            acc[et] = wmma_bf16(a, b, acc[et]);
        }
        __syncthreads();                              // done reading this buffer
    }

    float* mp = Mpart + (size_t)blockIdx.x * (DDIM * DDIM); // stored transposed: [e][d]
    #pragma unroll
    for (int et = 0; et < 8; ++et) {
        int e = et * 16 + ln;
        #pragma unroll
        for (int r = 0; r < 8; ++r) {
            int d = d0 + r + g * 8;
            mp[(size_t)e * DDIM + d] = acc[et][r];
        }
    }
}

// ---------------------------------------------------------------------------
// Kernel 4: deterministic KSPLIT-way reduction of M partials; fold in
// sqrt(128) and emit M^T in bf16 (B-operand layout for the final GEMM).
// ---------------------------------------------------------------------------
__global__ __launch_bounds__(256) void reduce_m_kernel(const float* __restrict__ Mpart,
                                                       unsigned short* __restrict__ MTbf) {
    int i = blockIdx.x * 256 + threadIdx.x;           // < 16384
    float s = 0.0f;
    #pragma unroll
    for (int b = 0; b < KSPLIT; ++b) s += Mpart[(size_t)b * (DDIM * DDIM) + i];
    MTbf[i] = f2bf(s * 11.313708498984761f);          // * sqrt(128)
}

// ---------------------------------------------------------------------------
// Kernel 5: out = q @ (scale*M).  512 WGs x 16 rows; wave w -> e-tile 16*w.
// K = 128 -> 4 WMMA steps. M^T bf16 (32 KB) stays L2-resident.
// ---------------------------------------------------------------------------
__global__ __launch_bounds__(256) void out_kernel(const unsigned short* __restrict__ q,
                                                  const unsigned short* __restrict__ MTbf,
                                                  float* __restrict__ out) {
    const int m0   = blockIdx.x * 16;
    const int wave = threadIdx.x >> 5;
    const int lane = threadIdx.x & 31;
    const int ln   = lane & 15;
    const int g    = lane >> 4;
    const int e0   = wave * 16;

    floatx8 acc = (floatx8)0.0f;
    #pragma unroll
    for (int s = 0; s < 4; ++s) {
        Frag a;                                       // A[m][d] = q row-major
        const unsigned short* qp = q + (size_t)(m0 + ln) * DDIM + s * 32;
        a.u.lo = *(const U128*)(qp + g * 16);
        a.u.hi = *(const U128*)(qp + g * 16 + 8);
        Frag b;                                       // B[d][e] = MTbf[e][d]
        const unsigned short* mp = MTbf + (size_t)(e0 + ln) * DDIM + s * 32 + g * 16;
        b.u.lo = *(const U128*)mp;
        b.u.hi = *(const U128*)(mp + 8);
        acc = wmma_bf16(a, b, acc);
    }

    const int col = e0 + ln;
    #pragma unroll
    for (int r = 0; r < 8; ++r) {
        int row = m0 + r + g * 8;
        out[(size_t)row * DDIM + col] = acc[r];
    }
}

// ---------------------------------------------------------------------------
// Workspace layout (bytes), total ~9.2 MB
// ---------------------------------------------------------------------------
static const size_t OFF_WBF   = 0;                                  // 786432
static const size_t OFF_Q     = OFF_WBF   + 3u * DDIM * EMBDIM * 2; // 2 MB
static const size_t OFF_KT    = OFF_Q     + (size_t)NROWS * DDIM * 2;
static const size_t OFF_VT    = OFF_KT    + (size_t)NROWS * DDIM * 2;
static const size_t OFF_MPART = OFF_VT    + (size_t)NROWS * DDIM * 2;   // KSPLIT*64KB
static const size_t OFF_MTBF  = OFF_MPART + (size_t)KSPLIT * DDIM * DDIM * 4;

extern "C" void kernel_launch(void* const* d_in, const int* in_sizes, int n_in,
                              void* d_out, int out_size, void* d_ws, size_t ws_size,
                              hipStream_t stream) {
    (void)in_sizes; (void)n_in; (void)out_size; (void)ws_size;

    const float* x  = (const float*)d_in[0];
    const float* Wq = (const float*)d_in[1];
    const float* bq = (const float*)d_in[2];
    const float* Wk = (const float*)d_in[3];
    const float* bk = (const float*)d_in[4];
    const float* Wv = (const float*)d_in[5];
    const float* bv = (const float*)d_in[6];
    float* out = (float*)d_out;

    char* ws = (char*)d_ws;
    unsigned short* Wbf   = (unsigned short*)(ws + OFF_WBF);
    unsigned short* qbf   = (unsigned short*)(ws + OFF_Q);
    unsigned short* kT    = (unsigned short*)(ws + OFF_KT);
    unsigned short* vT    = (unsigned short*)(ws + OFF_VT);
    float*          Mpart = (float*)         (ws + OFF_MPART);
    unsigned short* MTbf  = (unsigned short*)(ws + OFF_MTBF);

    prep_kernel    <<<  1536, 256, 0, stream>>>(Wq, Wk, Wv, Wbf);
    qkv_kernel     <<<   512, 256, 0, stream>>>(x, Wbf, bq, bk, bv, qbf, kT, vT);
    kv_kernel      <<<KSPLIT, 256, 0, stream>>>(kT, vT, Mpart);
    reduce_m_kernel<<<    64, 256, 0, stream>>>(Mpart, MTbf);
    out_kernel     <<<   512, 256, 0, stream>>>(qbf, MTbf, out);
}
